// ALiBiAttention_71210557768289
// MI455X (gfx1250) — matrix operand order repaired and checked by reading, not compiled, not run on hardware
//
#include <hip/hip_runtime.h>
#include <hip/hip_bf16.h>
#include <math.h>

// ---- problem constants (fixed by the reference) ----
#define BN 2
#define TT 2048
#define DM 1024
#define NH 16
#define HD 64
#define MROWS (BN * TT)   // 4096

typedef __attribute__((ext_vector_type(16))) _Float16 v16h;
typedef __attribute__((ext_vector_type(8)))  _Float16 v8h;
typedef __attribute__((ext_vector_type(8)))  float    v8f;

__device__ __forceinline__ v16h make16(v8h lo, v8h hi) {
    union { v8h h[2]; v16h v; } u;
    u.h[0] = lo; u.h[1] = hi;
    return u.v;
}

// A-fragment (16xK=32, f16, row-major source, leading dim ld):
// lane<16 : row = row0+lane,    K = k0..k0+15   (contiguous)
// lane>=16: row = row0+lane-16, K = k0+16..k0+31 (contiguous)
__device__ __forceinline__ v16h load_a16(const _Float16* base, int ld, int row0, int k0) {
    const int lane = threadIdx.x & 31;
    const int half = lane >> 4;
    const int row  = row0 + (lane & 15);
    const _Float16* p = base + (size_t)row * ld + k0 + half * 16;
    v8h lo = *(const v8h*)p;
    v8h hi = *(const v8h*)(p + 8);
    return make16(lo, hi);
}

// B-fragment (K=32 x 16). Source holds B^T row-major (ld): element B[k][n] = src[n*ld + k].
// lane<16 : N = n0+lane,    K = k0..k0+15   (contiguous)
// lane>=16: N = n0+lane-16, K = k0+16..k0+31 (contiguous)
__device__ __forceinline__ v16h load_b16(const _Float16* base, int ld, int n0, int k0) {
    const int lane = threadIdx.x & 31;
    const int half = lane >> 4;
    const int col  = n0 + (lane & 15);
    const _Float16* p = base + (size_t)col * ld + k0 + half * 16;
    v8h lo = *(const v8h*)p;
    v8h hi = *(const v8h*)(p + 8);
    return make16(lo, hi);
}

// ---------------- fp32 -> fp16 conversion ----------------
__global__ void cvt_f32_to_f16(const float* __restrict__ src, _Float16* __restrict__ dst, int n) {
    int i = blockIdx.x * blockDim.x + threadIdx.x;
    int stride = gridDim.x * blockDim.x;
    for (; i < n; i += stride) dst[i] = (_Float16)src[i];
}

// ---------------- GEMM: out = A(f16, MxDM) * W(f16, DMxDM)^T ----------------
// One wave computes a 32(M) x 64(N) tile: 2 A-fragments share 4 B-fragments
// per k-step => 8 WMMAs per 12 B128 loads.
// mode 0: out_h = f16, layout [B,H,T,HD]   (Q / K projections; scale folded in)
// mode 1: out_h = f16, layout [B,H,HD,T]   (V projection, pre-transposed for PV)
// mode 2: out_f = f32, layout row-major [M, DM] (final Wo projection)
__global__ __launch_bounds__(32) void gemm32x64(const _Float16* __restrict__ A,
                                                const _Float16* __restrict__ W,
                                                float scale, int mode,
                                                _Float16* __restrict__ out_h,
                                                float* __restrict__ out_f) {
    const int lane = threadIdx.x & 31;
    const int half = lane >> 4;
    const int lcol = lane & 15;
    const int m0 = blockIdx.x * 32;
    const int n0 = blockIdx.y * 64;

    v8f acc[2][4] = {};
    for (int k0 = 0; k0 < DM; k0 += 32) {
        v16h a0 = load_a16(A, DM, m0,      k0);
        v16h a1 = load_a16(A, DM, m0 + 16, k0);
#pragma unroll
        for (int nt = 0; nt < 4; ++nt) {
            v16h b = load_b16(W, DM, n0 + nt * 16, k0);
            acc[0][nt] = __builtin_amdgcn_wmma_f32_16x16x32_f16(
                false, a0, false, b, (short)0, acc[0][nt], false, false);
            acc[1][nt] = __builtin_amdgcn_wmma_f32_16x16x32_f16(
                false, a1, false, b, (short)0, acc[1][nt], false, false);
        }
    }

#pragma unroll
    for (int mt = 0; mt < 2; ++mt) {
#pragma unroll
        for (int nt = 0; nt < 4; ++nt) {
#pragma unroll
            for (int r = 0; r < 8; ++r) {
                const int row = m0 + mt * 16 + r + half * 8; // C layout: VGPR r -> M=r / 8+r
                const int col = n0 + nt * 16 + lcol;
                const float v = acc[mt][nt][r] * scale;
                if (mode == 2) {
                    out_f[(size_t)row * DM + col] = v;
                } else {
                    const int b = row >> 11;            // /T
                    const int t = row & (TT - 1);
                    const int hh = col >> 6;            // /HD
                    const int d  = col & (HD - 1);
                    const size_t idx = (mode == 0)
                        ? ((size_t)(b * NH + hh) * TT + t) * HD + d
                        : ((size_t)(b * NH + hh) * HD + d) * TT + t;
                    out_h[idx] = (_Float16)v;
                }
            }
        }
    }
}

// ---------------- Flash attention with ALiBi (causal) ----------------
// One wave handles a 32-row query tile (two 16-row WMMA tiles) for one (b,h),
// sharing each 32-key strip's K and V fragments across both query tiles.
// q,k : f16 [B,H,T,HD]   (q pre-scaled by 1/sqrt(HD))
// vt  : f16 [B,H,HD,T]   (transposed V)
// aout: f16 [B,T,DM]     (row-major, col = h*64+d) for the final projection
__global__ __launch_bounds__(32) void attn_alibi(const _Float16* __restrict__ q,
                                                 const _Float16* __restrict__ k,
                                                 const _Float16* __restrict__ vt,
                                                 _Float16* __restrict__ aout) {
    __shared__ __align__(16) _Float16 sP[2][16 * 32];

    const int lane = threadIdx.x & 31;
    const int half = lane >> 4;
    const int lcol = lane & 15;
    const int t0 = blockIdx.x * 32;
    const int h  = blockIdx.y;
    const int b  = blockIdx.z;

    const _Float16* Qb = q  + (size_t)(b * NH + h) * TT * HD;
    const _Float16* Kb = k  + (size_t)(b * NH + h) * TT * HD;
    const _Float16* Vb = vt + (size_t)(b * NH + h) * HD * TT;

    v16h aq[2][2];
#pragma unroll
    for (int tq = 0; tq < 2; ++tq) {
        aq[tq][0] = load_a16(Qb, HD, t0 + tq * 16, 0);    // d = 0..31
        aq[tq][1] = load_a16(Qb, HD, t0 + tq * 16, 32);   // d = 32..63
    }

    v8f o[2][4] = {};
    float mrow[2][8], lrow[2][8];
#pragma unroll
    for (int tq = 0; tq < 2; ++tq)
#pragma unroll
        for (int r = 0; r < 8; ++r) { mrow[tq][r] = -INFINITY; lrow[tq][r] = 0.0f; }

    // slope_h = 2^(-8*(h+1)/16) = 2^(-(h+1)/2)
    const float slope = exp2f(-0.5f * (float)(h + 1));

    for (int k0 = 0; k0 < t0 + 32; k0 += 32) {
        // ---- shared K fragments for keys [k0, k0+32) ----
        v16h b0a = load_b16(Kb, HD, k0,      0);
        v16h b0b = load_b16(Kb, HD, k0,      32);
        v16h b1a = load_b16(Kb, HD, k0 + 16, 0);
        v16h b1b = load_b16(Kb, HD, k0 + 16, 32);

        float fac[2][8];
        bool act[2];
#pragma unroll
        for (int tq = 0; tq < 2; ++tq) {
            const int tbase = t0 + tq * 16;
            act[tq] = (k0 < tbase + 16);          // strip fully masked for this tile?
            if (!act[tq]) continue;

            v8f s0 = {};
            s0 = __builtin_amdgcn_wmma_f32_16x16x32_f16(false, aq[tq][0], false, b0a, (short)0, s0, false, false);
            s0 = __builtin_amdgcn_wmma_f32_16x16x32_f16(false, aq[tq][1], false, b0b, (short)0, s0, false, false);
            v8f s1 = {};
            s1 = __builtin_amdgcn_wmma_f32_16x16x32_f16(false, aq[tq][0], false, b1a, (short)0, s1, false, false);
            s1 = __builtin_amdgcn_wmma_f32_16x16x32_f16(false, aq[tq][1], false, b1b, (short)0, s1, false, false);

            // ---- ALiBi bias + causal mask + online softmax ----
#pragma unroll
            for (int r = 0; r < 8; ++r) {
                const int qrow = tbase + r + half * 8;
                const int kc0 = k0 + lcol;
                const int kc1 = kc0 + 16;
                float x0 = (kc0 <= qrow) ? s0[r] - slope * (float)(qrow - kc0) : -INFINITY;
                float x1 = (kc1 <= qrow) ? s1[r] - slope * (float)(qrow - kc1) : -INFINITY;

                float tm = fmaxf(x0, x1);
#pragma unroll
                for (int off = 1; off < 16; off <<= 1)
                    tm = fmaxf(tm, __shfl_xor(tm, off, 32));

                const float nm = fmaxf(mrow[tq][r], tm);
                const float f  = expf(mrow[tq][r] - nm);
                const float p0 = expf(x0 - nm);
                const float p1 = expf(x1 - nm);

                float ps = p0 + p1;
#pragma unroll
                for (int off = 1; off < 16; off <<= 1)
                    ps += __shfl_xor(ps, off, 32);

                lrow[tq][r] = lrow[tq][r] * f + ps;
                mrow[tq][r] = nm;
                fac[tq][r]  = f;

                // stash P into LDS in plain [row][col] order for A-fragment reshape
                sP[tq][(r + half * 8) * 32 + lcol]      = (_Float16)p0;
                sP[tq][(r + half * 8) * 32 + 16 + lcol] = (_Float16)p1;
            }

            // rescale running output
#pragma unroll
            for (int dt = 0; dt < 4; ++dt)
#pragma unroll
                for (int r = 0; r < 8; ++r)
                    o[tq][dt][r] *= fac[tq][r];
        }

        // drain LDS stores before the fragment reloads
        asm volatile("s_wait_dscnt 0" ::: "memory");

        // ---- shared V fragments; O += P * V for each active tile ----
        v16h bv[4];
#pragma unroll
        for (int dt = 0; dt < 4; ++dt)
            bv[dt] = load_b16(Vb, TT, dt * 16, k0);   // V[k][d] contiguous via Vt

#pragma unroll
        for (int tq = 0; tq < 2; ++tq) {
            if (!act[tq]) continue;
            const v16h pf = load_a16(&sP[tq][0], 32, 0, 0);   // P as 16x32 A-fragment
#pragma unroll
            for (int dt = 0; dt < 4; ++dt)
                o[tq][dt] = __builtin_amdgcn_wmma_f32_16x16x32_f16(
                    false, pf, false, bv[dt], (short)0, o[tq][dt], false, false);
        }
    }

    // ---- normalize and emit f16 [B,T,DM] for the Wo GEMM ----
#pragma unroll
    for (int tq = 0; tq < 2; ++tq)
#pragma unroll
        for (int dt = 0; dt < 4; ++dt)
#pragma unroll
            for (int r = 0; r < 8; ++r) {
                const int t   = t0 + tq * 16 + r + half * 8;
                const int col = h * HD + dt * 16 + lcol;
                const float v = o[tq][dt][r] / lrow[tq][r];
                aout[(size_t)(b * TT + t) * DM + col] = (_Float16)v;
            }
}

extern "C" void kernel_launch(void* const* d_in, const int* in_sizes, int n_in,
                              void* d_out, int out_size, void* d_ws, size_t ws_size,
                              hipStream_t stream) {
    const float* x  = (const float*)d_in[0];
    const float* Wq = (const float*)d_in[1];
    const float* Wk = (const float*)d_in[2];
    const float* Wv = (const float*)d_in[3];
    const float* Wo = (const float*)d_in[4];
    float* out = (float*)d_out;

    char* ws = (char*)d_ws;
    size_t off = 0;
    _Float16* x16  = (_Float16*)(ws + off); off += (size_t)MROWS * DM * 2;   // 8 MB
    _Float16* wq16 = (_Float16*)(ws + off); off += (size_t)DM * DM * 2;      // 2 MB
    _Float16* wk16 = (_Float16*)(ws + off); off += (size_t)DM * DM * 2;
    _Float16* wv16 = (_Float16*)(ws + off); off += (size_t)DM * DM * 2;
    _Float16* wo16 = (_Float16*)(ws + off); off += (size_t)DM * DM * 2;
    _Float16* q16  = (_Float16*)(ws + off); off += (size_t)BN * NH * TT * HD * 2;  // 8 MB
    _Float16* k16  = (_Float16*)(ws + off); off += (size_t)BN * NH * TT * HD * 2;
    _Float16* vt16 = (_Float16*)(ws + off); off += (size_t)BN * NH * TT * HD * 2;
    _Float16* ao16 = (_Float16*)(ws + off); off += (size_t)MROWS * DM * 2;

    // 1) convert inputs to fp16
    {
        const int nx = MROWS * DM;
        const int nw = DM * DM;
        cvt_f32_to_f16<<<(nx + 255) / 256, 256, 0, stream>>>(x,  x16,  nx);
        cvt_f32_to_f16<<<(nw + 255) / 256, 256, 0, stream>>>(Wq, wq16, nw);
        cvt_f32_to_f16<<<(nw + 255) / 256, 256, 0, stream>>>(Wk, wk16, nw);
        cvt_f32_to_f16<<<(nw + 255) / 256, 256, 0, stream>>>(Wv, wv16, nw);
        cvt_f32_to_f16<<<(nw + 255) / 256, 256, 0, stream>>>(Wo, wo16, nw);
    }

    // 2) Q/K/V projections (Q pre-scaled by 1/sqrt(64); V pre-transposed)
    dim3 ggrid(MROWS / 32, DM / 64);
    gemm32x64<<<ggrid, 32, 0, stream>>>(x16, wq16, 0.125f, 0, q16, nullptr);
    gemm32x64<<<ggrid, 32, 0, stream>>>(x16, wk16, 1.0f,   0, k16, nullptr);
    gemm32x64<<<ggrid, 32, 0, stream>>>(x16, wv16, 1.0f,   1, vt16, nullptr);

    // 3) causal ALiBi flash attention (32 queries per wave)
    attn_alibi<<<dim3(TT / 32, NH, BN), 32, 0, stream>>>(q16, k16, vt16, ao16);

    // 4) output projection (fp32 out)
    gemm32x64<<<ggrid, 32, 0, stream>>>(ao16, wo16, 1.0f, 2, nullptr, out);
}
